// SelfCrossAttention_9603546874541
// MI455X (gfx1250) — matrix operand order repaired and checked by reading, not compiled, not run on hardware
//
#include <hip/hip_runtime.h>
#include <stdint.h>

typedef unsigned short u16;
typedef __bf16    v16bf __attribute__((ext_vector_type(16)));
typedef float     v8f   __attribute__((ext_vector_type(8)));
typedef unsigned  v4u   __attribute__((ext_vector_type(4)));
typedef int       v4i   __attribute__((ext_vector_type(4)));
typedef int       v8i   __attribute__((ext_vector_type(8)));

#define HW   16384
#define WIDE 128
#define CCH  512
#define K3C  1536
#define TOPK 42

union Frag {
  v16bf v;
  v4u   q[2];
  u16   h[16];
};

__device__ __forceinline__ u16 f2bf(float f) {
  unsigned u = __builtin_bit_cast(unsigned, f);
  unsigned r = u + 0x7FFFu + ((u >> 16) & 1u);
  return (u16)(r >> 16);
}
__device__ __forceinline__ float bf2f(u16 h) {
  return __builtin_bit_cast(float, (unsigned)h << 16);
}

// ---------------- K0: fp32 -> bf16 ----------------
__global__ __launch_bounds__(256) void cvt_bf16_kernel(const float* __restrict__ src,
                                                       u16* __restrict__ dst, int n) {
  int i = blockIdx.x * 256 + threadIdx.x;
  if (i < n) dst[i] = f2bf(src[i]);
}

// ---------------- K1/K5: bf16 GEMM + bias ----------------
// A: [M,K] bf16 row-major weights. X: [K,HW] bf16 activations (one batch).
// out: [M,HW] bf16 (OUT_F32=false) or fp32 (OUT_F32=true).
// Block tile: 128 M x 64 N, 8 waves, each wave = one 16-row strip x 4 n-tiles.
template <bool OUT_F32>
__global__ __launch_bounds__(256) void gemm_bias_kernel(
    const u16* __restrict__ A, const u16* __restrict__ X,
    const float* __restrict__ bias, void* __restrict__ outv, int M, int K) {
  __shared__ __align__(16) u16 ldsX[64 * 32];  // [n_local][k_local]
  const int tid = threadIdx.x;
  const int w = tid >> 5, l = tid & 31;
  const int lhalf = l >> 4, llow = l & 15;
  const int n0 = blockIdx.x * 64;
  const int m0 = blockIdx.y * 128;

  v8f acc[4] = {};
  const int mrow = m0 + w * 16 + llow;
  const u16* Arow = A + (size_t)mrow * K + 8 * lhalf;

  for (int kk = 0; kk < K; kk += 32) {
    __syncthreads();
    {  // stage X[kk..kk+31][n0..n0+63] transposed into LDS
      int ks = tid >> 3;         // 0..31
      int ns = (tid & 7) * 8;    // 0..56
      v4u d = *(const v4u*)(X + (size_t)(kk + ks) * HW + n0 + ns);
      const u16* hp = (const u16*)&d;
      #pragma unroll
      for (int j = 0; j < 8; ++j) ldsX[(ns + j) * 32 + ks] = hp[j];
    }
    __syncthreads();

    Frag a;  // A-frag: K = kk + 8*lhalf + {0..7} and {16..23}
    a.q[0] = *(const v4u*)(Arow + kk);
    a.q[1] = *(const v4u*)(Arow + kk + 16);
    __builtin_prefetch(Arow + kk + 64, 0, 3);

    #pragma unroll
    for (int j = 0; j < 4; ++j) {
      Frag bf;  // B-frag: col = j*16+llow, K = 8*lhalf + {0..7} and {16..23}
      const u16* bp = &ldsX[(j * 16 + llow) * 32 + 8 * lhalf];
      bf.q[0] = *(const v4u*)(bp);
      bf.q[1] = *(const v4u*)(bp + 16);
      acc[j] = __builtin_amdgcn_wmma_f32_16x16x32_bf16(
          false, a.v, false, bf.v, (short)0, acc[j], false, false);
    }
  }

  #pragma unroll
  for (int j = 0; j < 4; ++j) {
    int ncol = n0 + j * 16 + llow;
    #pragma unroll
    for (int d = 0; d < 8; ++d) {
      int m = m0 + w * 16 + d + 8 * lhalf;
      float val = acc[j][d] + bias[m];
      if (OUT_F32) ((float*)outv)[(size_t)m * HW + ncol] = val;
      else         ((u16*)outv)[(size_t)m * HW + ncol] = f2bf(val);
    }
  }
}

// ---------------- K2: depthwise 3x3, pad 1 ----------------
__global__ __launch_bounds__(256) void dwconv_kernel(
    const u16* __restrict__ in, const float* __restrict__ wdw,
    const float* __restrict__ bdw, u16* __restrict__ out) {
  int idx = blockIdx.x * 256 + threadIdx.x;  // < 1536*HW
  int ch = idx >> 14;
  int p  = idx & (HW - 1);
  int y = p >> 7, x = p & 127;
  float acc = bdw[ch];
  const u16* base = in + (size_t)ch * HW;
  const float* wp = wdw + ch * 9;
  #pragma unroll
  for (int ky = 0; ky < 3; ++ky) {
    int yy = y + ky - 1;
    if (yy < 0 || yy > 127) continue;
    #pragma unroll
    for (int kx = 0; kx < 3; ++kx) {
      int xx = x + kx - 1;
      if (xx < 0 || xx > 127) continue;
      acc += bf2f(base[yy * WIDE + xx]) * wp[ky * 3 + kx];
    }
  }
  out[idx] = f2bf(acc);
}

// ---------------- K3: 1/max(||row||,1e-12) over token dim ----------------
__global__ __launch_bounds__(256) void rownorm_kernel(
    const u16* __restrict__ qkv2, float* __restrict__ invq, float* __restrict__ invk) {
  __shared__ float red[256];
  int part = blockIdx.x >> 9;   // 0 = q rows, 1 = k rows
  int c = blockIdx.x & 511;
  const u16* row = qkv2 + (size_t)(part * CCH + c) * HW;
  float s = 0.f;
  for (int i = threadIdx.x; i < HW; i += 256) {
    float v = bf2f(row[i]);
    s += v * v;
  }
  red[threadIdx.x] = s;
  __syncthreads();
  for (int off = 128; off > 0; off >>= 1) {
    if (threadIdx.x < off) red[threadIdx.x] += red[threadIdx.x + off];
    __syncthreads();
  }
  if (threadIdx.x == 0)
    (part ? invk : invq)[c] = 1.0f / fmaxf(sqrtf(red[0]), 1e-12f);
}

// ---------------- K4: fused channel attention per head ----------------
// S = q.k^T (WMMA over 16384 tokens) -> scale by invq*invk*scale -> top-42
// threshold + softmax per row -> P (bf16, LDS) -> out = P @ V (WMMA), with
// V tiles DMA'd into LDS by the Tensor Data Mover (TENSOR_LOAD_TO_LDS).
__global__ __launch_bounds__(256) void attn_kernel(
    const u16* __restrict__ qkv2, const float* __restrict__ invq,
    const float* __restrict__ invk, const float* __restrict__ scale_p,
    u16* __restrict__ attout) {
  __shared__ __align__(16) float Sf[64 * 64];
  __shared__ __align__(16) u16   Pb[64 * 64];
  __shared__ __align__(16) u16   Vt[64 * 128];  // row-major [k][n_local], TDM dest

  const int h = blockIdx.x;
  const int tid = threadIdx.x;
  const int w = tid >> 5, l = tid & 31;
  const int lhalf = l >> 4, llow = l & 15;

  const u16* qbase = qkv2 + (size_t)(h * 64) * HW;
  const u16* kbase = qkv2 + (size_t)(CCH + h * 64) * HW;
  const u16* vbase = qkv2 + (size_t)(2 * CCH + h * 64) * HW;

  // -- Phase 1: raw 64x64 Gram matrix over token dim --
  const int mt = w >> 1;
  const int nt0 = (w & 1) * 2;
  v8f acc0 = {}, acc1 = {};
  const u16* aRow  = qbase + (size_t)(mt * 16 + llow) * HW + 8 * lhalf;
  const u16* b0Row = kbase + (size_t)(nt0 * 16 + llow) * HW + 8 * lhalf;
  const u16* b1Row = b0Row + (size_t)16 * HW;
  for (int kk = 0; kk < HW; kk += 32) {
    Frag a, b0, b1;
    a.q[0]  = *(const v4u*)(aRow + kk);
    a.q[1]  = *(const v4u*)(aRow + kk + 16);
    b0.q[0] = *(const v4u*)(b0Row + kk);
    b0.q[1] = *(const v4u*)(b0Row + kk + 16);
    b1.q[0] = *(const v4u*)(b1Row + kk);
    b1.q[1] = *(const v4u*)(b1Row + kk + 16);
    acc0 = __builtin_amdgcn_wmma_f32_16x16x32_bf16(false, a.v, false, b0.v, (short)0, acc0, false, false);
    acc1 = __builtin_amdgcn_wmma_f32_16x16x32_bf16(false, a.v, false, b1.v, (short)0, acc1, false, false);
  }

  // -- Phase 2: fold in L2 normalization + learned temperature --
  {
    float sc = scale_p[0];
    #pragma unroll
    for (int d = 0; d < 8; ++d) {
      int m = mt * 16 + d + 8 * lhalf;
      float iq = invq[h * 64 + m] * sc;
      int n0c = nt0 * 16 + llow;
      Sf[m * 64 + n0c]      = acc0[d] * iq * invk[h * 64 + n0c];
      Sf[m * 64 + n0c + 16] = acc1[d] * iq * invk[h * 64 + n0c + 16];
    }
  }
  __syncthreads();

  // -- Phase 3: top-42 threshold (rank selection by counting) + softmax --
  if (tid < 64) {
    const float* row = &Sf[tid * 64];
    float th = -3.4e38f;
    for (int j = 0; j < 64; ++j) {
      float vj = row[j];
      int cnt = 0;
      for (int i = 0; i < 64; ++i) cnt += (row[i] >= vj) ? 1 : 0;
      if (cnt >= TOPK && vj > th) th = vj;
    }
    float mx = -3.4e38f;
    for (int j = 0; j < 64; ++j) if (row[j] >= th) mx = fmaxf(mx, row[j]);
    float den = 0.f;
    for (int j = 0; j < 64; ++j) if (row[j] >= th) den += __expf(row[j] - mx);
    float rden = 1.0f / den;
    for (int j = 0; j < 64; ++j) {
      float p = (row[j] >= th) ? __expf(row[j] - mx) * rden : 0.f;
      Pb[tid * 64 + j] = f2bf(p);
    }
  }
  __syncthreads();

  // -- Phase 4: out = P @ V, V tiles DMA'd via Tensor Data Mover --
  const int mt4 = w & 3;
  Frag a0, a1;  // P rows, K chunks 0..31 and 32..63
  {
    const u16* pr = &Pb[(mt4 * 16 + llow) * 64 + 8 * lhalf];
    a0.q[0] = *(const v4u*)(pr);
    a0.q[1] = *(const v4u*)(pr + 16);
    a1.q[0] = *(const v4u*)(pr + 32);
    a1.q[1] = *(const v4u*)(pr + 48);
  }
  const unsigned ldsVt = (unsigned)(uintptr_t)(void*)&Vt[0];
  const int nb = (w >> 2) * 4;
  for (int c0 = 0; c0 < HW; c0 += 128) {
    __syncthreads();  // previous chunk fully consumed
    if (w == 0) {
      // TDM D#: 2-D tile, 128 bf16 cols x 64 rows, row stride HW elements.
      unsigned long long ga = (unsigned long long)(uintptr_t)(vbase + c0);
      v4u g0;
      g0[0] = 1u;                                   // count=1, user descriptor
      g0[1] = ldsVt;                                // lds_addr
      g0[2] = (unsigned)(ga & 0xFFFFFFFFu);         // global_addr[31:0]
      g0[3] = (unsigned)((ga >> 32) & 0x1FFFFFFu)   // global_addr[56:32]
              | (2u << 30);                         // type = 2 ("image")
      v8i g1;
      g1[0] = 0x00010000;                           // data_size = 2 bytes
      g1[1] = (int)(16384u << 16);                  // tensor_dim0 = 16384 (lo16)
      g1[2] = (int)(64u << 16);                     // tensor_dim0 hi; tensor_dim1 = 64 (lo16)
      g1[3] = (int)(128u << 16);                    // tensor_dim1 hi; tile_dim0 = 128
      g1[4] = 64;                                   // tile_dim1 = 64, tile_dim2 = 0
      g1[5] = 16384;                                // tensor_dim0_stride lo32
      g1[6] = 0;                                    // stride hi; dim1_stride lo
      g1[7] = 0;
      v4i gz = {0, 0, 0, 0};
#if defined(__clang_major__) && (__clang_major__ >= 23)
      v8i gz8 = {0, 0, 0, 0, 0, 0, 0, 0};
      __builtin_amdgcn_tensor_load_to_lds(g0, g1, gz, gz, gz8, 0);
#else
      __builtin_amdgcn_tensor_load_to_lds(g0, g1, gz, gz, 0);
#endif
      __builtin_amdgcn_s_wait_tensorcnt(0);
    }
    __syncthreads();  // publish TDM-written LDS to all waves

    #pragma unroll
    for (int j = 0; j < 4; ++j) {
      int nloc = (nb + j) * 16 + llow;
      Frag bA, bB;  // B-frags gathered column-wise from row-major Vt
      #pragma unroll
      for (int e = 0; e < 16; ++e) {
        bA.h[e] = Vt[(8 * lhalf + (e & 7) + ((e >> 3) << 4)) * 128 + nloc];
        bB.h[e] = Vt[(32 + 8 * lhalf + (e & 7) + ((e >> 3) << 4)) * 128 + nloc];
      }
      v8f acc = {};
      acc = __builtin_amdgcn_wmma_f32_16x16x32_bf16(false, a0.v, false, bA.v, (short)0, acc, false, false);
      acc = __builtin_amdgcn_wmma_f32_16x16x32_bf16(false, a1.v, false, bB.v, (short)0, acc, false, false);
      #pragma unroll
      for (int d = 0; d < 8; ++d) {
        int m = mt4 * 16 + d + 8 * lhalf;
        attout[(size_t)(h * 64 + m) * HW + c0 + (nb + j) * 16 + llow] = f2bf(acc[d]);
      }
    }
  }
}

extern "C" void kernel_launch(void* const* d_in, const int* in_sizes, int n_in,
                              void* d_out, int out_size, void* d_ws, size_t ws_size,
                              hipStream_t stream) {
  const float* x      = (const float*)d_in[0];
  const float* w_qkv  = (const float*)d_in[1];
  const float* b_qkv  = (const float*)d_in[2];
  const float* w_dw   = (const float*)d_in[3];
  const float* b_dw   = (const float*)d_in[4];
  const float* scale  = (const float*)d_in[5];
  const float* w_proj = (const float*)d_in[6];
  const float* b_proj = (const float*)d_in[7];

  char* ws = (char*)d_ws;
  size_t off = 0;
  auto wsalloc = [&](size_t bytes) -> void* {
    void* p = ws + off;
    off += (bytes + 255) & ~(size_t)255;
    return p;
  };
  u16* wqkvb  = (u16*)wsalloc((size_t)K3C * CCH * sizeof(u16));
  u16* wprojb = (u16*)wsalloc((size_t)CCH * CCH * sizeof(u16));
  u16* xb     = (u16*)wsalloc((size_t)CCH * HW * sizeof(u16));
  u16* qkv1   = (u16*)wsalloc((size_t)K3C * HW * sizeof(u16));
  u16* qkv2   = (u16*)wsalloc((size_t)K3C * HW * sizeof(u16));
  float* invq = (float*)wsalloc(CCH * sizeof(float));
  float* invk = (float*)wsalloc(CCH * sizeof(float));
  u16* attout = xb;  // xb is dead after the qkv GEMM; reuse for attention output

  cvt_bf16_kernel<<<K3C * CCH / 256, 256, 0, stream>>>(w_qkv, wqkvb, K3C * CCH);
  cvt_bf16_kernel<<<CCH * CCH / 256, 256, 0, stream>>>(w_proj, wprojb, CCH * CCH);

  for (int b = 0; b < 2; ++b) {
    const float* xsrc = x + (size_t)b * CCH * HW;
    cvt_bf16_kernel<<<CCH * HW / 256, 256, 0, stream>>>(xsrc, xb, CCH * HW);

    dim3 g1(HW / 64, K3C / 128);
    gemm_bias_kernel<false><<<g1, 256, 0, stream>>>(wqkvb, xb, b_qkv, qkv1, K3C, CCH);

    dwconv_kernel<<<(K3C * HW) / 256, 256, 0, stream>>>(qkv1, w_dw, b_dw, qkv2);

    rownorm_kernel<<<1024, 256, 0, stream>>>(qkv2, invq, invk);

    attn_kernel<<<8, 256, 0, stream>>>(qkv2, invq, invk, scale, attout);

    dim3 g2(HW / 64, CCH / 128);
    float* outb = (float*)d_out + (size_t)b * CCH * HW;
    gemm_bias_kernel<true><<<g2, 256, 0, stream>>>(wprojb, attout, b_proj, outb, CCH, CCH);
  }
  (void)in_sizes; (void)n_in; (void)out_size; (void)ws_size;
}
